// VNN_ResnetPointnet_4535485464857
// MI455X (gfx1250) — compile-verified, matrix operand order checked by reading, not run on hardware
//
#include <hip/hip_runtime.h>
#include <hip/hip_bf16.h>

typedef float v2f  __attribute__((ext_vector_type(2)));
typedef float v8f  __attribute__((ext_vector_type(8)));
typedef __bf16 v16bf __attribute__((ext_vector_type(16)));

#define EPSV 1e-6f
#define NPTS 4096
#define BATCH 4
#define KNN 20
#define NCOL (3 * NPTS)

// ---------------------------------------------------------------------------
// 1) kNN: per-thread query, LDS-tiled candidate points, insertion top-20
// ---------------------------------------------------------------------------
__global__ void __launch_bounds__(128) knn_kernel(const float* __restrict__ p,
                                                  int* __restrict__ knn) {
  __shared__ float tx[128], ty[128], tz[128];
  const int b = blockIdx.y;
  const int n = blockIdx.x * 128 + threadIdx.x;
  const long long qb = ((long long)b * NPTS + n) * 3;
  const float qx = p[qb], qy = p[qb + 1], qz = p[qb + 2];
  float best[KNN];
  int   bidx[KNN];
  for (int k = 0; k < KNN; ++k) { best[k] = -1e30f; bidx[k] = 0; }
  for (int t0 = 0; t0 < NPTS; t0 += 128) {
    const long long tb = ((long long)b * NPTS + t0 + threadIdx.x) * 3;
    tx[threadIdx.x] = p[tb]; ty[threadIdx.x] = p[tb + 1]; tz[threadIdx.x] = p[tb + 2];
    __syncthreads();
    for (int j = 0; j < 128; ++j) {
      const float dx = tx[j] - qx, dy = ty[j] - qy, dz = tz[j] - qz;
      const float nd = -(dx * dx + dy * dy + dz * dz);
      if (nd > best[KNN - 1]) {
        int pos = KNN - 1;
        while (pos > 0 && best[pos - 1] < nd) {
          best[pos] = best[pos - 1]; bidx[pos] = bidx[pos - 1]; --pos;
        }
        best[pos] = nd; bidx[pos] = t0 + j;
      }
    }
    __syncthreads();
  }
  for (int k = 0; k < KNN; ++k)
    knn[((long long)b * NPTS + n) * KNN + k] = bidx[k];
}

// ---------------------------------------------------------------------------
// 2) Fused: edge features -> VN conv(64x3)+lrelu -> VN maxpool over k (64x64)
//           -> fc_pos(256x64).  One block per point; avoids a 250MB tensor.
// ---------------------------------------------------------------------------
__global__ void __launch_bounds__(64) conv_pool_kernel(
    const float* __restrict__ p, const int* __restrict__ knn,
    const float* __restrict__ Wf, const float* __restrict__ Wd,
    const float* __restrict__ Wp, const float* __restrict__ Wfc,
    float* __restrict__ out, long long sOutb) {
  __shared__ float feat[3][3][KNN];   // [feature-ch][coord][k]
  __shared__ float net[64][3][KNN];
  __shared__ float pooled[64][3];
  const int b = blockIdx.y, n = blockIdx.x, o = threadIdx.x;
  const long long pbase = ((long long)b * NPTS + n) * 3;
  const float qx = p[pbase], qy = p[pbase + 1], qz = p[pbase + 2];
  if (o < KNN) {
    const int m = knn[((long long)b * NPTS + n) * KNN + o];
    const long long mb = ((long long)b * NPTS + m) * 3;
    const float nx = p[mb], ny = p[mb + 1], nz = p[mb + 2];
    feat[0][0][o] = nx - qx; feat[0][1][o] = ny - qy; feat[0][2][o] = nz - qz;
    feat[1][0][o] = qx;      feat[1][1][o] = qy;      feat[1][2][o] = qz;
    feat[2][0][o] = ny * qz - nz * qy;   // cross(neigh, x)
    feat[2][1][o] = nz * qx - nx * qz;
    feat[2][2][o] = nx * qy - ny * qx;
  }
  __syncthreads();
  const float wf0 = Wf[o * 3], wf1 = Wf[o * 3 + 1], wf2 = Wf[o * 3 + 2];
  const float wd0 = Wd[o * 3], wd1 = Wd[o * 3 + 1], wd2 = Wd[o * 3 + 2];
  for (int k = 0; k < KNN; ++k) {
    float pv[3], dv[3], dot = 0.f, dsq = 0.f;
    for (int cd = 0; cd < 3; ++cd) {
      const float f0 = feat[0][cd][k], f1 = feat[1][cd][k], f2 = feat[2][cd][k];
      pv[cd] = wf0 * f0 + wf1 * f1 + wf2 * f2;
      dv[cd] = wd0 * f0 + wd1 * f1 + wd2 * f2;
      dot += pv[cd] * dv[cd]; dsq += dv[cd] * dv[cd];
    }
    const float s = (dot >= 0.f) ? 0.f : dot / (dsq + EPSV);
    for (int cd = 0; cd < 3; ++cd) net[o][cd][k] = pv[cd] - s * dv[cd];
  }
  __syncthreads();
  // VN max pool over the 20 neighbors with direction W_pool (64x64)
  float bv = -1e30f; int bk = 0;
  for (int k = 0; k < KNN; ++k) {
    float d0 = 0.f, d1 = 0.f, d2 = 0.f;
    for (int c = 0; c < 64; ++c) {
      const float w = Wp[o * 64 + c];
      d0 += w * net[c][0][k]; d1 += w * net[c][1][k]; d2 += w * net[c][2][k];
    }
    const float dot = net[o][0][k] * d0 + net[o][1][k] * d1 + net[o][2][k] * d2;
    if (dot > bv) { bv = dot; bk = k; }
  }
  pooled[o][0] = net[o][0][bk]; pooled[o][1] = net[o][1][bk]; pooled[o][2] = net[o][2][bk];
  __syncthreads();
  // fc_pos: 256x64
  for (int j = 0; j < 4; ++j) {
    const int oo = j * 64 + o;
    float a0 = 0.f, a1 = 0.f, a2 = 0.f;
    for (int c = 0; c < 64; ++c) {
      const float w = Wfc[oo * 64 + c];
      a0 += w * pooled[c][0]; a1 += w * pooled[c][1]; a2 += w * pooled[c][2];
    }
    float* O = out + (long long)b * sOutb + (long long)(oo * 3) * NPTS + n;
    O[0] = a0; O[NPTS] = a1; O[2 * NPTS] = a2;
  }
}

// ---------------------------------------------------------------------------
// 3) Channel GEMM on the matrix pipe: Y[M x NCOL] = W[M x K] * X[K x NCOL] (+Z)
//    fp32 WMMA 16x16x4; one wave per 16x64 D-tile (4 accumulators share one
//    A fragment), 4 waves (64 rows) / block.  All 4 B fragments are loaded
//    before the 4 WMMAs so the compiler can clause the loads and stagger
//    s_wait_loadcnt instead of fully serializing load->wait->wmma.
// ---------------------------------------------------------------------------
__global__ void __launch_bounds__(128) gemm_wmma_f32(
    const float* __restrict__ W, const float* __restrict__ X,
    const float* __restrict__ Z, float* __restrict__ Y,
    int M, int K, int Ncols,
    long long sXb, long long sYb, long long sZb) {
  const int lane = threadIdx.x & 31;
  const int wave = threadIdx.x >> 5;
  const int b = blockIdx.z;
  const int mTile = blockIdx.y * 4 + wave;
  const int nBase = blockIdx.x * 64;
  if (mTile * 16 >= M) return;
  const float* Xb = X + (long long)b * sXb;
  float* Yb = Y + (long long)b * sYb;
  const int hi = lane >> 4;     // upper half-wave holds K+2/K+3 slices
  const int lo = lane & 15;
  const int m = mTile * 16 + lo;
  v8f acc[4] = {};
#if __has_builtin(__builtin_amdgcn_wmma_f32_16x16x4_f32)
  for (int k0 = 0; k0 < K; k0 += 4) {
    const int ka = k0 + 2 * hi;
    v2f a;
    a.x = W[(long long)m * K + ka];
    a.y = W[(long long)m * K + ka + 1];
    const float* r0 = Xb + (long long)ka * Ncols + nBase + lo;
    const float* r1 = r0 + Ncols;
    v2f bbv[4];
#pragma unroll
    for (int t = 0; t < 4; ++t) {
      bbv[t].x = r0[t * 16];
      bbv[t].y = r1[t * 16];
    }
#pragma unroll
    for (int t = 0; t < 4; ++t) {
      acc[t] = __builtin_amdgcn_wmma_f32_16x16x4_f32(false, a, false, bbv[t],
                                                     (short)0, acc[t], false, false);
    }
  }
#else
  // Fallback: bf16 WMMA 16x16x32 (codegen-confirmed builtin), fp32 accumulate.
  for (int k0 = 0; k0 < K; k0 += 32) {
    v16bf a;
#pragma unroll
    for (int j = 0; j < 16; ++j) {
      const int kk = k0 + j + 8 * hi + ((j >= 8) ? 8 : 0);
      a[j] = (__bf16)W[(long long)m * K + kk];
    }
    v16bf bbv[4];
#pragma unroll
    for (int t = 0; t < 4; ++t)
#pragma unroll
      for (int j = 0; j < 16; ++j) {
        const int kk = k0 + j + 8 * hi + ((j >= 8) ? 8 : 0);
        bbv[t][j] = (__bf16)Xb[(long long)kk * Ncols + nBase + t * 16 + lo];
      }
#pragma unroll
    for (int t = 0; t < 4; ++t)
      acc[t] = __builtin_amdgcn_wmma_f32_16x16x32_bf16(false, a, false, bbv[t],
                                                       (short)0, acc[t], false, false);
  }
#endif
#pragma unroll
  for (int t = 0; t < 4; ++t) {
#pragma unroll
    for (int i = 0; i < 8; ++i) {
      const int row = mTile * 16 + hi * 8 + i;
      const long long offo = (long long)row * Ncols + nBase + t * 16 + lo;
      float v = acc[t][i];
      if (Z) v += Z[(long long)b * sZb + offo];
      Yb[offo] = v;
    }
  }
}

// ---------------------------------------------------------------------------
// 4) VN LeakyReLU (ns=0): act = (dot>=0) ? x : x - dot/(|d|^2+eps)*d
//    Writes activation in-place over D (one thread per (b,c,n) coord-triple).
// ---------------------------------------------------------------------------
__global__ void __launch_bounds__(256) lrelu_kernel(
    const float* __restrict__ X, float* __restrict__ D,
    int C, long long sXb, long long sDb) {
  const long long g = (long long)blockIdx.x * 256 + threadIdx.x;
  const long long total = (long long)BATCH * C * NPTS;
  if (g >= total) return;
  const int n = (int)(g % NPTS);
  const long long r = g / NPTS;
  const int c = (int)(r % C);
  const int b = (int)(r / C);
  const float* x = X + (long long)b * sXb + (long long)(c * 3) * NPTS + n;
  float* d = D + (long long)b * sDb + (long long)(c * 3) * NPTS + n;
  const float x0 = x[0], x1 = x[NPTS], x2 = x[2 * NPTS];
  const float d0 = d[0], d1 = d[NPTS], d2 = d[2 * NPTS];
  const float dot = x0 * d0 + x1 * d1 + x2 * d2;
  const float dsq = d0 * d0 + d1 * d1 + d2 * d2;
  const float s = (dot >= 0.f) ? 0.f : dot / (dsq + EPSV);
  d[0] = x0 - s * d0; d[NPTS] = x1 - s * d1; d[2 * NPTS] = x2 - s * d2;
}

// ---------------------------------------------------------------------------
// 5) VN max pool over N. mode 0: broadcast-fill the concat half (channels
//    128..255) of the ping-pong buffer (out pre-offset).  mode 1: compact.
// ---------------------------------------------------------------------------
__global__ void __launch_bounds__(256) pool_n_kernel(
    const float* __restrict__ X, const float* __restrict__ D,
    float* __restrict__ out, int mode,
    long long sXb, long long sDb, long long sOutb) {
  const int c = blockIdx.x, b = blockIdx.y, t = threadIdx.x;
  const float* Xb = X + (long long)b * sXb + (long long)(c * 3) * NPTS;
  const float* Db = D + (long long)b * sDb + (long long)(c * 3) * NPTS;
  float bv = -1e30f; int bn = 0;
  for (int n = t; n < NPTS; n += 256) {
    const float dot = Xb[n] * Db[n] + Xb[NPTS + n] * Db[NPTS + n] +
                      Xb[2 * NPTS + n] * Db[2 * NPTS + n];
    if (dot > bv) { bv = dot; bn = n; }
  }
  __shared__ float rv[256];
  __shared__ int ri[256];
  rv[t] = bv; ri[t] = bn;
  __syncthreads();
  for (int s = 128; s > 0; s >>= 1) {
    if (t < s) {
      if (rv[t + s] > rv[t] || (rv[t + s] == rv[t] && ri[t + s] < ri[t])) {
        rv[t] = rv[t + s]; ri[t] = ri[t + s];
      }
    }
    __syncthreads();
  }
  const int nb = ri[0];
  const float v0 = Xb[nb], v1 = Xb[NPTS + nb], v2 = Xb[2 * NPTS + nb];
  if (mode == 0) {
    float* O = out + (long long)b * sOutb + (long long)(c * 3) * NPTS;
    for (int n = t; n < NPTS; n += 256) {
      O[n] = v0; O[NPTS + n] = v1; O[2 * NPTS + n] = v2;
    }
  } else if (t == 0) {
    out[(long long)b * sOutb + c * 3] = v0;
    out[(long long)b * sOutb + c * 3 + 1] = v1;
    out[(long long)b * sOutb + c * 3 + 2] = v2;
  }
}

// ---------------------------------------------------------------------------
// 6) Tiny head: lrelu(pooled, actvn_c) then fc_c -> (B,128,3)
// ---------------------------------------------------------------------------
__global__ void __launch_bounds__(128) final_kernel(
    const float* __restrict__ pv, const float* __restrict__ Wa,
    const float* __restrict__ Wc, float* __restrict__ out) {
  __shared__ float xin[128][3];
  __shared__ float act[128][3];
  const int b = blockIdx.x, o = threadIdx.x;
  for (int cd = 0; cd < 3; ++cd) xin[o][cd] = pv[b * 384 + o * 3 + cd];
  __syncthreads();
  float d0 = 0.f, d1 = 0.f, d2 = 0.f;
  for (int c = 0; c < 128; ++c) {
    const float w = Wa[o * 128 + c];
    d0 += w * xin[c][0]; d1 += w * xin[c][1]; d2 += w * xin[c][2];
  }
  const float x0 = xin[o][0], x1 = xin[o][1], x2 = xin[o][2];
  const float dot = x0 * d0 + x1 * d1 + x2 * d2;
  const float dsq = d0 * d0 + d1 * d1 + d2 * d2;
  const float s = (dot >= 0.f) ? 0.f : dot / (dsq + EPSV);
  act[o][0] = x0 - s * d0; act[o][1] = x1 - s * d1; act[o][2] = x2 - s * d2;
  __syncthreads();
  float a0 = 0.f, a1 = 0.f, a2 = 0.f;
  for (int c = 0; c < 128; ++c) {
    const float w = Wc[o * 128 + c];
    a0 += w * act[c][0]; a1 += w * act[c][1]; a2 += w * act[c][2];
  }
  out[b * 384 + o * 3] = a0;
  out[b * 384 + o * 3 + 1] = a1;
  out[b * 384 + o * 3 + 2] = a2;
}

// ---------------------------------------------------------------------------
extern "C" void kernel_launch(void* const* d_in, const int* in_sizes, int n_in,
                              void* d_out, int out_size, void* d_ws, size_t ws_size,
                              hipStream_t stream) {
  (void)in_sizes; (void)n_in; (void)out_size; (void)ws_size;
  const long long S256 = (long long)256 * NCOL;
  const long long S128 = (long long)128 * NCOL;

  // setup_inputs() dict order: p, then params in insertion order.
  const float* p  = (const float*)d_in[0];
  const float* Wcf = (const float*)d_in[1];  // conv_pos_feat 64x3
  const float* Wcd = (const float*)d_in[2];  // conv_pos_dir  64x3
  const float* Wpp = (const float*)d_in[3];  // pool_pos      64x64
  const float* Wfp = (const float*)d_in[4];  // fc_pos        256x64
  const float* Wac = (const float*)d_in[5];  // actvn_c       128x128
  const float* Wfc = (const float*)d_in[6];  // fc_c          128x128
  const float *Wb_a0[5], *Wb_f0[5], *Wb_a1[5], *Wb_f1[5], *Wb_sc[5], *Wb_pl[5];
  for (int i = 0; i < 5; ++i) {
    Wb_a0[i] = (const float*)d_in[7 + 6 * i];   // actvn0 256x256
    Wb_f0[i] = (const float*)d_in[8 + 6 * i];   // fc0    128x256
    Wb_a1[i] = (const float*)d_in[9 + 6 * i];   // actvn1 128x128
    Wb_f1[i] = (const float*)d_in[10 + 6 * i];  // fc1    128x128
    Wb_sc[i] = (const float*)d_in[11 + 6 * i];  // shortcut 128x256
    Wb_pl[i] = (const float*)d_in[12 + 6 * i];  // pool_i 128x128
  }

  char* ws = (char*)d_ws;
  size_t off = 0;
  auto walloc = [&](size_t bytes) -> void* {
    void* q = ws + off;
    off = (off + bytes + 255) & ~(size_t)255;
    return q;
  };
  int*   knnIdx = (int*)walloc((size_t)BATCH * NPTS * KNN * sizeof(int));
  float* X0   = (float*)walloc((size_t)BATCH * S256 * sizeof(float));
  float* X1   = (float*)walloc((size_t)BATCH * S256 * sizeof(float));
  float* Dbuf = (float*)walloc((size_t)BATCH * S256 * sizeof(float));
  float* Net  = (float*)walloc((size_t)BATCH * S128 * sizeof(float));
  float* Net2 = (float*)walloc((size_t)BATCH * S128 * sizeof(float));
  float* Pvec = (float*)walloc((size_t)BATCH * 128 * 3 * sizeof(float));

  knn_kernel<<<dim3(NPTS / 128, BATCH), 128, 0, stream>>>(p, knnIdx);
  conv_pool_kernel<<<dim3(NPTS, BATCH), 64, 0, stream>>>(
      p, knnIdx, Wcf, Wcd, Wpp, Wfp, X0, S256);

  auto gemm = [&](const float* W, const float* X, const float* Z, float* Y,
                  int M, int K, long long sXb, long long sYb, long long sZb) {
    dim3 g(NCOL / 64, M / 64, BATCH);
    gemm_wmma_f32<<<g, 128, 0, stream>>>(W, X, Z, Y, M, K, NCOL, sXb, sYb, sZb);
  };
  auto lrelu = [&](const float* X, float* D, int C, long long sXb, long long sDb) {
    const long long tot = (long long)BATCH * C * NPTS;
    lrelu_kernel<<<dim3((unsigned)((tot + 255) / 256)), 256, 0, stream>>>(
        X, D, C, sXb, sDb);
  };
  auto resblock = [&](int i, const float* Xin, float* Yout) {
    gemm(Wb_a0[i], Xin, nullptr, Dbuf, 256, 256, S256, S256, 0);   // d = W_a0 x
    lrelu(Xin, Dbuf, 256, S256, S256);                             // act in Dbuf
    gemm(Wb_f0[i], Dbuf, nullptr, Net, 128, 256, S256, S128, 0);   // net = fc0
    gemm(Wb_a1[i], Net, nullptr, Dbuf, 128, 128, S128, S128, 0);   // d = W_a1 net
    lrelu(Net, Dbuf, 128, S128, S128);
    gemm(Wb_f1[i], Dbuf, nullptr, Net2, 128, 128, S128, S128, 0);  // dx = fc1
    gemm(Wb_sc[i], Xin, Net2, Yout, 128, 256, S256, S256, S128);   // sc(x)+dx
  };
  auto pool = [&](int i, float* Xn, int mode, float* outp, long long sOutb) {
    gemm(Wb_pl[i], Xn, nullptr, Dbuf, 128, 128, S256, S128, 0);
    pool_n_kernel<<<dim3(128, BATCH), 256, 0, stream>>>(
        Xn, Dbuf, outp, mode, S256, S128, sOutb);
  };

  // ping-pong 256-channel buffers; resblock writes ch 0..127, pool fills
  // ch 128..255 with the broadcast pooled vector (= concat for free)
  resblock(0, X0, X1);
  pool(0, X1, 0, X1 + (long long)128 * NCOL, S256);
  resblock(1, X1, X0);
  pool(1, X0, 0, X0 + (long long)128 * NCOL, S256);
  resblock(2, X0, X1);
  pool(2, X1, 0, X1 + (long long)128 * NCOL, S256);
  resblock(3, X1, X0);
  pool(3, X0, 0, X0 + (long long)128 * NCOL, S256);
  resblock(4, X0, X1);
  pool(4, X1, 1, Pvec, 384);

  final_kernel<<<BATCH, 128, 0, stream>>>(Pvec, Wac, Wfc, (float*)d_out);
}
